// FlowLMExportWrapper_76321568849978
// MI455X (gfx1250) — matrix operand order repaired and checked by reading, not compiled, not run on hardware
//
#include <hip/hip_runtime.h>
#include <hip/hip_bf16.h>
#include <math.h>

// ---------------------------------------------------------------------------
// Types for WMMA
// ---------------------------------------------------------------------------
typedef __attribute__((ext_vector_type(16))) _Float16 v16h;
typedef __attribute__((ext_vector_type(8)))  float    v8f;

union Frag { unsigned int u[8]; v16h v; };

__device__ __forceinline__ unsigned short f2h(float f) {
  _Float16 h = (_Float16)f;
  return __builtin_bit_cast(unsigned short, h);
}

__device__ __forceinline__ float gelu_tanh(float x) {
  return 0.5f * x * (1.0f + tanhf(0.7978845608028654f * (x + 0.044715f * x * x * x)));
}
__device__ __forceinline__ float silu_f(float x) {
  return x / (1.0f + expf(-x));
}
__device__ __forceinline__ float wave_sum(float v) {
  #pragma unroll
  for (int off = 16; off > 0; off >>= 1) v += __shfl_xor(v, off, 32);
  return v;
}

// ---------------------------------------------------------------------------
// Tiled GEMM:  C[M,N] (+)= act(A[M,K] @ B[K,N] + bias[N])
// A : f16 activations in global (produced by upstream kernels).
// B : f32 weights in global, converted f32->f16 while staging into LDS
//     (weights are consumed exactly once per launch, so converting on the fly
//      is strictly cheaper than a pre-conversion pass).
// OT: output type (float, or _Float16 when feeding the next GEMM).
// Requires M%128==0, N%128==0, K%32==0 (true for every call site here).
// ACT: 0=none, 1=gelu(tanh). RES: C += result (only with OT=float).
// All global staging addresses are maintained as incremented pointers so the
// inner loop carries no 64-bit address recomputation.
// ---------------------------------------------------------------------------
#define BM 128
#define BN 128
#define BK 32

template<int ACT, bool RES, typename OT>
__global__ __launch_bounds__(256)
void gemm_wmma(const _Float16* __restrict__ A, const float* __restrict__ B,
               const float* __restrict__ bias, OT* __restrict__ C,
               int M, int N, int K) {
  __shared__ unsigned short lsA[BM * BK];   // [row][k] contiguous K-pairs (f16)
  __shared__ unsigned short lsB[BK * BN];   // packed: u32 (k>>1)*BN + col = (K even, K odd)

  const int tid  = threadIdx.x;
  const int lane = tid & 31;
  const int wave = tid >> 5;
  const int half = lane >> 4;
  const int l16  = lane & 15;
  const int wm   = wave >> 2;            // 0..1  (64-row slab)
  const int wn   = wave & 3;             // 0..3  (32-col slab)
  const int bm0  = blockIdx.y * BM;
  const int bn0  = blockIdx.x * BN;
  const int K2   = K >> 1;               // row length of A in u32 units

  v8f acc[4][2];
  #pragma unroll
  for (int mt = 0; mt < 4; ++mt)
    #pragma unroll
    for (int nt = 0; nt < 2; ++nt)
      acc[mt][nt] = v8f{};

  unsigned int*       lsA32w = (unsigned int*)lsA;
  const unsigned int* lsA32  = (const unsigned int*)lsA;
  const unsigned int* lsB32  = (const unsigned int*)lsB;

  // ---- per-thread staging base pointers (advanced each K-step) ------------
  // A: thread owns u32 column (tid&15) of rows (tid>>4) + 16*i, i=0..7
  const unsigned int* A32 = (const unsigned int*)A;
  const unsigned int* ap  = A32 + (size_t)(bm0 + (tid >> 4)) * K2 + (tid & 15);
  const size_t aRowStep   = (size_t)16 * K2;          // 16 rows down
  // B: thread owns 16B segment at col bn0+(tid&31)*4 of rows (tid>>5) + 8*i
  const float* bp = B + (size_t)(tid >> 5) * N + bn0 + (tid & 31) * 4;
  const size_t bRowStep = (size_t)8 * N;              // 8 rows down
  const int cB  = (tid & 31) * 4;                     // LDS col
  const int loB = (tid >> 5) & 1;                     // K parity (const: rows step by 8)
  const int rhB = tid >> 6;                           // (row>>1) for i=0
  // prefetch pointers (one cacheline probe per thread; A rows / B segments)
  const unsigned int* pfA = A32 + (size_t)(bm0 + (tid & 127)) * K2 + (BK >> 1);
  const float*        pfB = B + (size_t)(BK + ((tid & 127) >> 2)) * N + bn0 + (tid & 3) * 32;

  for (int k0 = 0; k0 < K; k0 += BK) {
    // ---- stage A tile (128x32 f16) : raw u32 copy, layout matches LDS
    #pragma unroll
    for (int i = 0; i < 8; ++i)
      lsA32w[tid + i * 256] = ap[(size_t)i * aRowStep];
    // ---- stage B tile (32x128), f32 -> f16, K-pair-packed layout
    #pragma unroll
    for (int i = 0; i < 4; ++i) {
      float4 v4 = *(const float4*)(bp + (size_t)i * bRowStep);
      unsigned short* d = &lsB[((rhB + i * 4) * BN + cB) * 2 + loB];
      d[0] = f2h(v4.x);
      d[2] = f2h(v4.y);
      d[4] = f2h(v4.z);
      d[6] = f2h(v4.w);
    }
    // ---- prefetch next K-tile (global_prefetch_b8)
    if (k0 + BK < K) {
      if (tid < 128) __builtin_prefetch(pfA, 0, 3);
      else           __builtin_prefetch(pfB, 0, 3);
    }
    ap  += BK >> 1;               // 32 f16 = 16 u32
    bp  += (size_t)BK * N;
    pfA += BK >> 1;
    pfB += (size_t)BK * N;
    __syncthreads();

    // ---- load fragments per ISA 7.12.2 layouts
    Frag af[4], bf[2];
    #pragma unroll
    for (int mt = 0; mt < 4; ++mt) {
      int row = wm * 64 + mt * 16 + l16;
      #pragma unroll
      for (int j = 0; j < 8; ++j) {
        // u32 j holds K pair: j<4 -> K = half*8 + 2j ; j>=4 -> K = 16 + half*8 + 2(j-4)
        int idx = (j < 4) ? (half * 4 + j) : (4 + half * 4 + j);
        af[mt].u[j] = lsA32[row * 16 + idx];
      }
    }
    #pragma unroll
    for (int nt = 0; nt < 2; ++nt) {
      int col = wn * 32 + nt * 16 + l16;
      #pragma unroll
      for (int j = 0; j < 8; ++j) {
        // u32 j holds K pair: j<4 -> K = half*8 + 2j ; j>=4 -> K = 16 + half*8 + 2(j-4)
        bf[nt].u[j] = lsB32[((j < 4) ? (half * 4 + j) : (4 + half * 4 + j)) * BN + col];
      }
    }

    #pragma unroll
    for (int mt = 0; mt < 4; ++mt)
      #pragma unroll
      for (int nt = 0; nt < 2; ++nt)
        acc[mt][nt] = __builtin_amdgcn_wmma_f32_16x16x32_f16(
            false, af[mt].v, false, bf[nt].v, (short)0, acc[mt][nt], false, false);

    __syncthreads();
  }

  // ---- epilogue: bias + activation (+ residual)
  #pragma unroll
  for (int mt = 0; mt < 4; ++mt) {
    #pragma unroll
    for (int nt = 0; nt < 2; ++nt) {
      int gc = bn0 + wn * 32 + nt * 16 + l16;
      float bb = bias[gc];
      #pragma unroll
      for (int i = 0; i < 8; ++i) {
        int gr = bm0 + wm * 64 + mt * 16 + half * 8 + i;
        float v = acc[mt][nt][i] + bb;
        if (ACT == 1) v = gelu_tanh(v);
        size_t o = (size_t)gr * N + gc;
        if constexpr (RES) v += (float)C[o];
        C[o] = (OT)v;
      }
    }
  }
}

// ---------------------------------------------------------------------------
// Constants for this model
// ---------------------------------------------------------------------------
#define BATCH 8
#define TTXT  64
#define TSEQ  256
#define TTOT  320
#define DIMM  1024
#define LDIMM 128
#define HEADS_N 16
#define HDIM_N  64
#define FFN_N   4096
#define HID_N   1024
#define NSTEPS  16

// ---------------------------------------------------------------------------
// Elementwise / prep kernels
// ---------------------------------------------------------------------------
__global__ void fix_seq_kernel(const float* __restrict__ seq,
                               const float* __restrict__ bos,
                               _Float16* __restrict__ out, int n) {
  int i = blockIdx.x * blockDim.x + threadIdx.x;
  if (i >= n) return;
  float v = seq[i];
  out[i] = (_Float16)(isnan(v) ? bos[i & (LDIMM - 1)] : v);
}

__global__ void copy_text_kernel(const float* __restrict__ text,
                                 float* __restrict__ h, int n) {
  int i = blockIdx.x * blockDim.x + threadIdx.x;
  if (i >= n) return;
  int d   = i & (DIMM - 1);
  int row = i >> 10;            // b*64 + t
  int b   = row >> 6;
  int t   = row & 63;
  h[((size_t)(b * TTOT + t) * DIMM) + d] = text[i];
}

__global__ void scatter_x_kernel(const float* __restrict__ x,
                                 float* __restrict__ h, int n) {
  int i = blockIdx.x * blockDim.x + threadIdx.x;
  if (i >= n) return;
  int d   = i & (DIMM - 1);
  int row = i >> 10;            // b*256 + t
  int b   = row >> 8;
  int t   = row & 255;
  h[((size_t)(b * TTOT + TTXT + t) * DIMM) + d] = x[i];
}

// ---------------------------------------------------------------------------
// LayerNorm: one wave per row (D=1024); f32 in, f16 out (GEMM A operand)
// ---------------------------------------------------------------------------
__global__ __launch_bounds__(256)
void ln_kernel(const float* __restrict__ x, const float* __restrict__ g,
               const float* __restrict__ bta, _Float16* __restrict__ out, int rows) {
  int wave = threadIdx.x >> 5;
  int lane = threadIdx.x & 31;
  int row  = blockIdx.x * 8 + wave;
  if (row >= rows) return;
  const float* xr = x + (size_t)row * DIMM;
  float s = 0.f, ss = 0.f;
  for (int i = lane; i < DIMM; i += 32) {
    float v = xr[i];
    s += v; ss += v * v;
  }
  s  = wave_sum(s);
  ss = wave_sum(ss);
  float mean = s * (1.0f / DIMM);
  float var  = ss * (1.0f / DIMM) - mean * mean;
  float rs   = rsqrtf(var + 1e-5f);
  _Float16* orow = out + (size_t)row * DIMM;
  for (int i = lane; i < DIMM; i += 32)
    orow[i] = (_Float16)((xr[i] - mean) * rs * g[i] + bta[i]);
}

// LayerNorm over only the last token of each batch (8 rows), f32 out
__global__ __launch_bounds__(256)
void ln_last_kernel(const float* __restrict__ h, const float* __restrict__ g,
                    const float* __restrict__ bta, float* __restrict__ last) {
  int b    = threadIdx.x >> 5;   // 8 waves, one per batch
  int lane = threadIdx.x & 31;
  const float* xr = h + (size_t)(b * TTOT + (TTOT - 1)) * DIMM;
  float s = 0.f, ss = 0.f;
  for (int i = lane; i < DIMM; i += 32) { float v = xr[i]; s += v; ss += v * v; }
  s  = wave_sum(s);
  ss = wave_sum(ss);
  float mean = s * (1.0f / DIMM);
  float var  = ss * (1.0f / DIMM) - mean * mean;
  float rs   = rsqrtf(var + 1e-5f);
  for (int i = lane; i < DIMM; i += 32)
    last[(size_t)b * DIMM + i] = (xr[i] - mean) * rs * g[i] + bta[i];
}

// EOS head: out_eos[b] = (last[b] . W_eos + b_eos) > 0.5
__global__ __launch_bounds__(256)
void eos_kernel(const float* __restrict__ last, const float* __restrict__ Weos,
                const float* __restrict__ beos, float* __restrict__ out) {
  int b    = threadIdx.x >> 5;
  int lane = threadIdx.x & 31;
  float s = 0.f;
  for (int i = lane; i < DIMM; i += 32)
    s += last[(size_t)b * DIMM + i] * Weos[i];
  s = wave_sum(s);
  if (lane == 0)
    out[b] = ((s + beos[0]) > 0.5f) ? 1.0f : 0.0f;
}

// ---------------------------------------------------------------------------
// Causal attention, streaming softmax; one wave per (b, head, query row).
// qkv layout: row = b*320+t, 3072 cols = [q | k | v], each head h at h*64.
// Output written f16 (it feeds the O-projection GEMM).
// ---------------------------------------------------------------------------
__global__ __launch_bounds__(256)
void attn_kernel(const float* __restrict__ qkv, _Float16* __restrict__ o) {
  int gw   = blockIdx.x * 8 + (threadIdx.x >> 5);
  int lane = threadIdx.x & 31;
  int b    = gw / (HEADS_N * TTOT);
  int rem  = gw % (HEADS_N * TTOT);
  int hd   = rem / TTOT;
  int qi   = rem % TTOT;

  const float* base = qkv + (size_t)(b * TTOT) * 3072;
  const float* qr   = base + (size_t)qi * 3072 + hd * HDIM_N;
  int d2 = lane * 2;
  float q0 = qr[d2], q1 = qr[d2 + 1];

  float m = -1e30f, l = 0.f, a0 = 0.f, a1 = 0.f;
  for (int j = 0; j <= qi; ++j) {
    const float* kr = base + (size_t)j * 3072 + DIMM + hd * HDIM_N;
    float part = q0 * kr[d2] + q1 * kr[d2 + 1];
    float s = wave_sum(part) * 0.125f;           // 1/sqrt(64)
    float mn   = fmaxf(m, s);
    float corr = expf(m - mn);
    float p    = expf(s - mn);
    const float* vr = base + (size_t)j * 3072 + 2 * DIMM + hd * HDIM_N;
    l  = l  * corr + p;
    a0 = a0 * corr + p * vr[d2];
    a1 = a1 * corr + p * vr[d2 + 1];
    m = mn;
  }
  float inv = 1.0f / l;
  _Float16* orow = o + (size_t)(b * TTOT + qi) * DIMM + hd * HDIM_N;
  orow[d2]     = (_Float16)(a0 * inv);
  orow[d2 + 1] = (_Float16)(a1 * inv);
}

// ---------------------------------------------------------------------------
// Flow-matching MLP steps (tiny: 8 rows). One thread per output element.
// inp = [ last(1024), s, t, current(128) ]  (K = 1154)
// ---------------------------------------------------------------------------
__global__ void flow1_kernel(const float* __restrict__ last,
                             const float* __restrict__ cur,
                             const float* __restrict__ W,   // (1154,1024)
                             const float* __restrict__ bias,
                             float* __restrict__ f1, float sv, float tv) {
  int gid = blockIdx.x * blockDim.x + threadIdx.x;   // b*1024 + j
  if (gid >= BATCH * HID_N) return;
  int b = gid >> 10, j = gid & (HID_N - 1);
  float acc = bias[j];
  const float* lb = last + (size_t)b * DIMM;
  for (int c = 0; c < DIMM; ++c) acc += lb[c] * W[(size_t)c * HID_N + j];
  acc += sv * W[(size_t)1024 * HID_N + j];
  acc += tv * W[(size_t)1025 * HID_N + j];
  const float* cb = cur + (size_t)b * LDIMM;
  for (int c = 0; c < LDIMM; ++c) acc += cb[c] * W[(size_t)(1026 + c) * HID_N + j];
  f1[gid] = silu_f(acc);
}

__global__ void flow2_kernel(const float* __restrict__ f1,
                             const float* __restrict__ W,   // (1024,1024)
                             const float* __restrict__ bias,
                             float* __restrict__ f2) {
  int gid = blockIdx.x * blockDim.x + threadIdx.x;
  if (gid >= BATCH * HID_N) return;
  int b = gid >> 10, j = gid & (HID_N - 1);
  float acc = bias[j];
  const float* fb = f1 + (size_t)b * HID_N;
  for (int c = 0; c < HID_N; ++c) acc += fb[c] * W[(size_t)c * HID_N + j];
  f2[gid] = silu_f(acc);
}

__global__ void flow3_kernel(const float* __restrict__ f2,
                             const float* __restrict__ W,   // (1024,128)
                             const float* __restrict__ bias,
                             float* __restrict__ cur) {
  int gid = blockIdx.x * blockDim.x + threadIdx.x;   // b*128 + j
  if (gid >= BATCH * LDIMM) return;
  int b = gid >> 7, j = gid & (LDIMM - 1);
  float acc = bias[j];
  const float* fb = f2 + (size_t)b * HID_N;
  for (int c = 0; c < HID_N; ++c) acc += fb[c] * W[(size_t)c * LDIMM + j];
  cur[gid] += acc * (1.0f / NSTEPS);
}

__global__ void copy_kernel(const float* __restrict__ src, float* __restrict__ dst, int n) {
  int i = blockIdx.x * blockDim.x + threadIdx.x;
  if (i < n) dst[i] = src[i];
}

// ---------------------------------------------------------------------------
// Host-side orchestration
// ---------------------------------------------------------------------------
extern "C" void kernel_launch(void* const* d_in, const int* in_sizes, int n_in,
                              void* d_out, int out_size, void* d_ws, size_t ws_size,
                              hipStream_t stream) {
  (void)in_sizes; (void)n_in; (void)out_size; (void)ws_size;

  const float* seq    = (const float*)d_in[0];
  const float* text   = (const float*)d_in[1];
  const float* noise  = (const float*)d_in[2];
  /* d_in[3] = lsd_decode_steps (traced int, == 16) */
  const float* bos    = (const float*)d_in[4];
  const float* W_in   = (const float*)d_in[5];
  const float* b_in   = (const float*)d_in[6];
  const float* ln1_g  = (const float*)d_in[7];
  const float* ln1_b  = (const float*)d_in[8];
  const float* Wqkv   = (const float*)d_in[9];
  const float* bqkv   = (const float*)d_in[10];
  const float* Wo     = (const float*)d_in[11];
  const float* bo     = (const float*)d_in[12];
  const float* ln2_g  = (const float*)d_in[13];
  const float* ln2_b  = (const float*)d_in[14];
  const float* W1     = (const float*)d_in[15];
  const float* b1     = (const float*)d_in[16];
  const float* W2     = (const float*)d_in[17];
  const float* b2     = (const float*)d_in[18];
  const float* onormg = (const float*)d_in[19];
  const float* onormb = (const float*)d_in[20];
  const float* W_eos  = (const float*)d_in[21];
  const float* b_eos  = (const float*)d_in[22];
  const float* Wf1    = (const float*)d_in[23];
  const float* bf1    = (const float*)d_in[24];
  const float* Wf2    = (const float*)d_in[25];
  const float* bf2    = (const float*)d_in[26];
  const float* Wf3    = (const float*)d_in[27];
  const float* bf3    = (const float*)d_in[28];

  float* out = (float*)d_out;

  // Workspace layout (regions sized in f32 units; f16 regions use half but
  // keep the generous f32-sized spacing for simplicity/alignment)
  float* ws   = (float*)d_ws;
  float*     h    = ws;                                 // 2560*1024 f32
  _Float16*  z    = (_Float16*)(h    + (size_t)2560 * 1024);  // 2560*1024 f16
  float*     qkv  = (float*)   ((float*)z + (size_t)2560 * 1024);
  _Float16*  ao   = (_Float16*)(qkv  + (size_t)2560 * 3072);  // 2560*1024 f16
  _Float16*  ffn  = (_Float16*)((float*)ao + (size_t)2560 * 1024); // 2560*4096 f16
  float*     xtmp = (float*)   ((float*)ffn + (size_t)2560 * 4096);
  _Float16*  seqf = (_Float16*)(xtmp + (size_t)2048 * 1024);  // 2048*128 f16
  float*     last = (float*)   ((float*)seqf + (size_t)2048 * 128);
  float*     cur  = last + 8 * 1024;
  float*     f1b  = cur  + 8 * 128;
  float*     f2b  = f1b  + 8 * 1024;

  const int TPB = 256;
  const int M   = BATCH * TTOT;     // 2560

  // --- input embedding -----------------------------------------------------
  {
    int n = BATCH * TSEQ * LDIMM;   // 262144
    fix_seq_kernel<<<(n + TPB - 1) / TPB, TPB, 0, stream>>>(seq, bos, seqf, n);
  }
  {
    int n = BATCH * TTXT * DIMM;    // 524288
    copy_text_kernel<<<(n + TPB - 1) / TPB, TPB, 0, stream>>>(text, h, n);
  }
  gemm_wmma<0, false, float>
      <<<dim3(DIMM / BN, (BATCH * TSEQ) / BM), TPB, 0, stream>>>(
      seqf, W_in, b_in, xtmp, BATCH * TSEQ, DIMM, LDIMM);
  {
    int n = BATCH * TSEQ * DIMM;    // 2097152
    scatter_x_kernel<<<(n + TPB - 1) / TPB, TPB, 0, stream>>>(xtmp, h, n);
  }

  // --- transformer layers --------------------------------------------------
  for (int l = 0; l < 4; ++l) {
    const float* Wqkv_l = Wqkv + (size_t)l * DIMM * 3 * DIMM;
    const float* bqkv_l = bqkv + (size_t)l * 3 * DIMM;
    const float* Wo_l   = Wo   + (size_t)l * DIMM * DIMM;
    const float* bo_l   = bo   + (size_t)l * DIMM;
    const float* W1_l   = W1   + (size_t)l * DIMM * FFN_N;
    const float* b1_l   = b1   + (size_t)l * FFN_N;
    const float* W2_l   = W2   + (size_t)l * FFN_N * DIMM;
    const float* b2_l   = b2   + (size_t)l * DIMM;

    ln_kernel<<<M / 8, TPB, 0, stream>>>(h, ln1_g + l * DIMM, ln1_b + l * DIMM, z, M);
    gemm_wmma<0, false, float>
        <<<dim3(3 * DIMM / BN, M / BM), TPB, 0, stream>>>(
        z, Wqkv_l, bqkv_l, qkv, M, 3 * DIMM, DIMM);
    attn_kernel<<<(BATCH * HEADS_N * TTOT) / 8, TPB, 0, stream>>>(qkv, ao);
    gemm_wmma<0, true, float>
        <<<dim3(DIMM / BN, M / BM), TPB, 0, stream>>>(
        ao, Wo_l, bo_l, h, M, DIMM, DIMM);

    ln_kernel<<<M / 8, TPB, 0, stream>>>(h, ln2_g + l * DIMM, ln2_b + l * DIMM, z, M);
    gemm_wmma<1, false, _Float16>
        <<<dim3(FFN_N / BN, M / BM), TPB, 0, stream>>>(
        z, W1_l, b1_l, ffn, M, FFN_N, DIMM);
    gemm_wmma<0, true, float>
        <<<dim3(DIMM / BN, M / BM), TPB, 0, stream>>>(
        ffn, W2_l, b2_l, h, M, DIMM, FFN_N);
  }

  // --- output head (only final token per batch is consumed) ---------------
  ln_last_kernel<<<1, TPB, 0, stream>>>(h, onormg, onormb, last);
  eos_kernel<<<1, TPB, 0, stream>>>(last, W_eos, b_eos, out + BATCH * LDIMM);

  // --- flow decoder (16 sequential steps, TEMP=1 so cur = noise) -----------
  copy_kernel<<<(BATCH * LDIMM + TPB - 1) / TPB, TPB, 0, stream>>>(noise, cur, BATCH * LDIMM);
  for (int i = 0; i < NSTEPS; ++i) {
    float sv = (float)i / NSTEPS;
    float tv = (float)(i + 1) / NSTEPS;
    flow1_kernel<<<(BATCH * HID_N) / TPB, TPB, 0, stream>>>(last, cur, Wf1, bf1, f1b, sv, tv);
    flow2_kernel<<<(BATCH * HID_N) / TPB, TPB, 0, stream>>>(f1b, Wf2, bf2, f2b);
    flow3_kernel<<<(BATCH * LDIMM + TPB - 1) / TPB, TPB, 0, stream>>>(f2b, Wf3, bf3, cur);
  }
  copy_kernel<<<(BATCH * LDIMM + TPB - 1) / TPB, TPB, 0, stream>>>(cur, out, BATCH * LDIMM);
}
